// Attention_27522150433460
// MI455X (gfx1250) — matrix operand order repaired and checked by reading, not compiled, not run on hardware
//
#include <hip/hip_runtime.h>
#include <hip/hip_bf16.h>

typedef __bf16 bf16;
typedef __attribute__((ext_vector_type(16))) __bf16 v16bf;
typedef __attribute__((ext_vector_type(8)))  __bf16 v8bf;
typedef __attribute__((ext_vector_type(8)))  float  v8f;

#define B_ 8
#define N_ 1024
#define C_ 768
#define H_ 12
#define D_ 64
#define SCALE_ 0.125f

// round-to-nearest-even f32 -> bf16 (bit-level)
__device__ __forceinline__ bf16 f2bf(float f) {
    unsigned u = __builtin_bit_cast(unsigned, f);
    u += 0x7FFFu + ((u >> 16) & 1u);
    unsigned short s = (unsigned short)(u >> 16);
    return __builtin_bit_cast(bf16, s);
}

// A-operand (16xK row tile, row-major, leading dim ldk):
//  lane m = lane%16, half = lane/16; element e -> K = (e/8)*16 + half*8 + (e%8)
__device__ __forceinline__ v16bf load_A(const bf16* base, int ldk, int m, int half, int k0) {
    const v8bf* p0 = (const v8bf*)(base + (size_t)m * ldk + k0 + half * 8);
    const v8bf* p1 = (const v8bf*)(base + (size_t)m * ldk + k0 + 16 + half * 8);
    v8bf lo = *p0, hi = *p1;
    v16bf a;
#pragma unroll
    for (int i = 0; i < 8; ++i) { a[i] = lo[i]; a[i + 8] = hi[i]; }
    return a;
}

// B-operand (Kx16 tile, column n contiguous over K with stride ldk):
//  lane n = lane%16, half = lane/16; element e -> K = (e/8)*16 + half*8 + (e%8)
__device__ __forceinline__ v16bf load_B(const bf16* base, int ldk, int n, int half, int k0) {
    const bf16* p = base + (size_t)n * ldk + k0 + half * 8;
    v8bf lo = *(const v8bf*)p;
    v8bf hi = *(const v8bf*)(p + 16);
    v16bf b;
#pragma unroll
    for (int i = 0; i < 8; ++i) { b[i] = lo[i]; b[i + 8] = hi[i]; }
    return b;
}

__device__ __forceinline__ v8f wmma_bf16(v16bf a, v16bf b, v8f c) {
    return __builtin_amdgcn_wmma_f32_16x16x32_bf16(false, a, false, b, (short)0, c,
                                                   false, false);
}

// ---------------------------------------------------------------- cast kernel
__global__ void cast_kernel(const float* __restrict__ src, bf16* __restrict__ dst, int n) {
    int i = blockIdx.x * blockDim.x + threadIdx.x;
    if (i < n) dst[i] = f2bf(src[i]);
}

// ------------------------------------------------------- QKV projection GEMM
// Wave tile: 32 rows x 64 cols = 8 accumulators; 6 operand loads / 8 WMMAs per k-step.
// MODE 0: Q (scaled, [B,H,N,D]); 1: K ([B,H,N,D]); 2: V transposed ([B,H,D,N]).
template <int MODE>
__global__ void qkv_gemm(const bf16* __restrict__ X, const bf16* __restrict__ W,
                         bf16* __restrict__ Out) {
    const int lane = threadIdx.x & 31;
    const int wave = threadIdx.x >> 5;
    const int half = lane >> 4;
    const int l16  = lane & 15;
    const int m0 = (blockIdx.x * 4 + wave) * 32;  // 32-token tile per wave
    const int n0 = blockIdx.y * 64;               // 64-channel tile per wave
    v8f zero = {};
    v8f acc[2][4];
#pragma unroll
    for (int mi = 0; mi < 2; ++mi)
#pragma unroll
        for (int ni = 0; ni < 4; ++ni) acc[mi][ni] = zero;

    for (int k0 = 0; k0 < C_; k0 += 32) {
        if (k0 + 64 < C_) {  // stream next slabs via global_prefetch_b8
            __builtin_prefetch(X + (size_t)(m0 + lane) * C_ + k0 + 64, 0, 1);
            __builtin_prefetch(W + (size_t)(n0 + lane) * C_ + k0 + 64, 0, 1);
            __builtin_prefetch(W + (size_t)(n0 + 32 + lane) * C_ + k0 + 64, 0, 1);
        }
        v16bf a0 = load_A(X + (size_t)m0 * C_, C_, l16, half, k0);
        v16bf a1 = load_A(X + (size_t)(m0 + 16) * C_, C_, l16, half, k0);
#pragma unroll
        for (int ni = 0; ni < 4; ++ni) {
            v16bf b = load_B(W + (size_t)(n0 + ni * 16) * C_, C_, l16, half, k0);
            acc[0][ni] = wmma_bf16(a0, b, acc[0][ni]);
            acc[1][ni] = wmma_bf16(a1, b, acc[1][ni]);
        }
    }
#pragma unroll
    for (int mi = 0; mi < 2; ++mi)
#pragma unroll
        for (int ni = 0; ni < 4; ++ni) {
            if (MODE == 2) {
                // V transposed: consecutive e -> consecutive token -> packed store
                int row0 = m0 + mi * 16 + 8 * half;       // first of 8 tokens
                int col  = n0 + ni * 16 + l16;            // output channel
                int bb = row0 >> 10, nn0 = row0 & 1023;
                int hh = col >> 6,  dd = col & 63;
                v8bf pk;
#pragma unroll
                for (int e = 0; e < 8; ++e) pk[e] = f2bf(acc[mi][ni][e]);
                *(v8bf*)(Out + (((size_t)bb * H_ + hh) * D_ + dd) * N_ + nn0) = pk;
            } else {
#pragma unroll
                for (int e = 0; e < 8; ++e) {
                    int row = m0 + mi * 16 + e + 8 * half;
                    int col = n0 + ni * 16 + l16;
                    int bb = row >> 10, nn = row & 1023;
                    int hh = col >> 6,  dd = col & 63;
                    float v = acc[mi][ni][e];
                    Out[(((size_t)bb * H_ + hh) * N_ + nn) * D_ + dd] =
                        f2bf(MODE == 0 ? v * SCALE_ : v);
                }
            }
        }
}

// ----------------------------------------------------- flash attention kernel
// Computes S^T = K * Q^T so each lane owns ONE query row:
//   scores C-layout: key = e + 8*half (within 16-key tile), query = lane%16.
// Softmax stats are in-lane folds + one xor-16 shuffle. PV is done transposed:
//   O^T = V^T * P^T with A = Vt (row-major [d][key]) and B = P^T from LDS.
__global__ void attn_kernel(const bf16* __restrict__ Q, const bf16* __restrict__ K,
                            const bf16* __restrict__ Vt, const float* __restrict__ bias,
                            bf16* __restrict__ O) {
    __shared__ __align__(16) bf16 ldsP[4][16 * 32];   // per-wave P^T: [query][key]
    const int lane = threadIdx.x & 31;
    const int wave = threadIdx.x >> 5;
    const int half = lane >> 4;
    const int l16  = lane & 15;
    const int bh = blockIdx.x;               // 0..B*H-1
    const int h  = bh % H_;
    const int b  = bh / H_;
    const int r0 = blockIdx.y * 64 + wave * 16;  // this wave's 16 query rows

    const bf16* Qbh = Q  + (size_t)bh * N_ * D_;
    const bf16* Kbh = K  + (size_t)bh * N_ * D_;
    const bf16* Vbh = Vt + (size_t)bh * D_ * N_;
    const float* brow = bias + (size_t)h * N_ * N_ + (size_t)(r0 + l16) * N_;

    // Q rows as B-operand (column = query), hoisted; D=64 -> two K=32 slabs
    v16bf qb0 = load_B(Qbh, D_, r0 + l16, half, 0);
    v16bf qb1 = load_B(Qbh, D_, r0 + l16, half, 32);

    float m = -1e30f, l = 0.0f;
    v8f zero = {};
    v8f o_acc[4];
#pragma unroll
    for (int t = 0; t < 4; ++t) o_acc[t] = zero;

    bf16* P = ldsP[wave];

    for (int key0 = 0; key0 < N_; key0 += 32) {
        // ---- S^T: two 16-key tiles; A = K rows (keys), B = Q columns (queries)
        v8f s[2];
#pragma unroll
        for (int t = 0; t < 2; ++t) {
            v16bf ka0 = load_A(Kbh + (size_t)(key0 + t * 16) * D_, D_, l16, half, 0);
            v16bf ka1 = load_A(Kbh + (size_t)(key0 + t * 16) * D_, D_, l16, half, 32);
            v8f c = zero;
            c = wmma_bf16(ka0, qb0, c);
            c = wmma_bf16(ka1, qb1, c);
            // bias: element e -> key = key0 + t*16 + 8*half + e (contiguous!)
            const float* bp = brow + key0 + t * 16 + 8 * half;
            float4 b0 = *(const float4*)bp;
            float4 b1 = *(const float4*)(bp + 4);
            c[0] += b0.x; c[1] += b0.y; c[2] += b0.z; c[3] += b0.w;
            c[4] += b1.x; c[5] += b1.y; c[6] += b1.z; c[7] += b1.w;
            s[t] = c;
        }
        // ---- online softmax stats (per-lane scalars: one query per lane)
        float mx = fmaxf(s[0][0], s[1][0]);
#pragma unroll
        for (int e = 1; e < 8; ++e) mx = fmaxf(mx, fmaxf(s[0][e], s[1][e]));
        mx = fmaxf(mx, __shfl_xor(mx, 16, 32));   // partner half holds other 16 keys
        float nm = fmaxf(m, mx);
        float sc = __expf(m - nm);
        m = nm;
        l *= sc;
#pragma unroll
        for (int t = 0; t < 4; ++t)
#pragma unroll
            for (int e = 0; e < 8; ++e) o_acc[t][e] *= sc;
        // ---- P = exp(s - m): packed bf16 -> LDS [query][key], sum in-lane
        float ts = 0.0f;
#pragma unroll
        for (int t = 0; t < 2; ++t) {
            v8bf pk;
#pragma unroll
            for (int e = 0; e < 8; ++e) {
                float p = __expf(s[t][e] - nm);
                ts += p;
                pk[e] = f2bf(p);
            }
            *(v8bf*)(P + l16 * 32 + t * 16 + 8 * half) = pk;   // ds_store_b128
        }
        ts += __shfl_xor(ts, 16, 32);
        l += ts;
        asm volatile("s_wait_dscnt 0" ::: "memory");  // same-wave LDS RAW fence
        // ---- O^T += V^T @ P^T : A = Vt rows (d), B = P^T columns (queries)
        v16bf pb = load_B(P, 32, l16, half, 0);       // ds_load_b128 x2
#pragma unroll
        for (int t = 0; t < 4; ++t) {
            v16bf va = load_A(Vbh + (size_t)(t * 16) * N_, N_, l16, half, key0);
            o_acc[t] = wmma_bf16(va, pb, o_acc[t]);
        }
    }
    // ---- normalize; element e -> d = t*16 + 8*half + e (contiguous channels)
    float inv = 1.0f / l;
    const int row = r0 + l16;
#pragma unroll
    for (int t = 0; t < 4; ++t) {
        v8bf pk;
#pragma unroll
        for (int e = 0; e < 8; ++e) pk[e] = f2bf(o_acc[t][e] * inv);
        *(v8bf*)(O + ((size_t)b * N_ + row) * C_ + h * D_ + t * 16 + 8 * half) = pk;
    }
}

// ------------------------------------------------------ output projection GEMM
__global__ void proj_gemm(const bf16* __restrict__ X, const bf16* __restrict__ W,
                          const float* __restrict__ bias, float* __restrict__ Out) {
    const int lane = threadIdx.x & 31;
    const int wave = threadIdx.x >> 5;
    const int half = lane >> 4;
    const int l16  = lane & 15;
    const int m0 = (blockIdx.x * 4 + wave) * 32;
    const int n0 = blockIdx.y * 64;
    v8f zero = {};
    v8f acc[2][4];
#pragma unroll
    for (int mi = 0; mi < 2; ++mi)
#pragma unroll
        for (int ni = 0; ni < 4; ++ni) acc[mi][ni] = zero;

    for (int k0 = 0; k0 < C_; k0 += 32) {
        if (k0 + 64 < C_) {
            __builtin_prefetch(X + (size_t)(m0 + lane) * C_ + k0 + 64, 0, 1);
            __builtin_prefetch(W + (size_t)(n0 + lane) * C_ + k0 + 64, 0, 1);
            __builtin_prefetch(W + (size_t)(n0 + 32 + lane) * C_ + k0 + 64, 0, 1);
        }
        v16bf a0 = load_A(X + (size_t)m0 * C_, C_, l16, half, k0);
        v16bf a1 = load_A(X + (size_t)(m0 + 16) * C_, C_, l16, half, k0);
#pragma unroll
        for (int ni = 0; ni < 4; ++ni) {
            v16bf b = load_B(W + (size_t)(n0 + ni * 16) * C_, C_, l16, half, k0);
            acc[0][ni] = wmma_bf16(a0, b, acc[0][ni]);
            acc[1][ni] = wmma_bf16(a1, b, acc[1][ni]);
        }
    }
#pragma unroll
    for (int mi = 0; mi < 2; ++mi)
#pragma unroll
        for (int ni = 0; ni < 4; ++ni)
#pragma unroll
            for (int e = 0; e < 8; ++e) {
                int row = m0 + mi * 16 + e + 8 * half;
                int col = n0 + ni * 16 + l16;
                Out[(size_t)row * C_ + col] = acc[mi][ni][e] + bias[col];
            }
}

extern "C" void kernel_launch(void* const* d_in, const int* in_sizes, int n_in,
                              void* d_out, int out_size, void* d_ws, size_t ws_size,
                              hipStream_t stream) {
    const float* x  = (const float*)d_in[0];
    const float* ab = (const float*)d_in[1];
    const float* Wq = (const float*)d_in[2];
    const float* Wk = (const float*)d_in[3];
    const float* Wv = (const float*)d_in[4];
    const float* Wp = (const float*)d_in[5];
    const float* bp = (const float*)d_in[6];
    float* out = (float*)d_out;
    (void)in_sizes; (void)n_in; (void)out_size; (void)ws_size;

    const size_t nx = (size_t)B_ * N_ * C_;   // 6,291,456 (also == B*H*N*D)
    const size_t nw = (size_t)C_ * C_;        //   589,824
    bf16* ws    = (bf16*)d_ws;
    bf16* x_bf  = ws;  ws += nx;
    bf16* wq_bf = ws;  ws += nw;
    bf16* wk_bf = ws;  ws += nw;
    bf16* wv_bf = ws;  ws += nw;
    bf16* wp_bf = ws;  ws += nw;
    bf16* Qb    = ws;  ws += nx;   // [B,H,N,D], pre-scaled
    bf16* Kb    = ws;  ws += nx;   // [B,H,N,D]
    bf16* Vtb   = ws;  ws += nx;   // [B,H,D,N]
    bf16* AOb   = ws;  ws += nx;   // [B*N, C]

    const int CT = 256;
    cast_kernel<<<(int)((nx + CT - 1) / CT), CT, 0, stream>>>(x,  x_bf,  (int)nx);
    cast_kernel<<<(int)((nw + CT - 1) / CT), CT, 0, stream>>>(Wq, wq_bf, (int)nw);
    cast_kernel<<<(int)((nw + CT - 1) / CT), CT, 0, stream>>>(Wk, wk_bf, (int)nw);
    cast_kernel<<<(int)((nw + CT - 1) / CT), CT, 0, stream>>>(Wv, wv_bf, (int)nw);
    cast_kernel<<<(int)((nw + CT - 1) / CT), CT, 0, stream>>>(Wp, wp_bf, (int)nw);

    // 8192 rows / (4 waves * 32 rows) = 64 blocks; 768 cols / 64 = 12
    dim3 gg(64, 12);
    qkv_gemm<0><<<gg, 128, 0, stream>>>(x_bf, wq_bf, Qb);
    qkv_gemm<1><<<gg, 128, 0, stream>>>(x_bf, wk_bf, Kb);
    qkv_gemm<2><<<gg, 128, 0, stream>>>(x_bf, wv_bf, Vtb);

    attn_kernel<<<dim3(B_ * H_, N_ / 64), 128, 0, stream>>>(Qb, Kb, Vtb, ab, AOb);

    proj_gemm<<<gg, 128, 0, stream>>>(AOb, wp_bf, bp, out);
}
